// BloomAttentionWithSeparatedQKV_33543694581925
// MI455X (gfx1250) — matrix operand order repaired and checked by reading, not compiled, not run on hardware
//
#include <hip/hip_runtime.h>

#define HIDDEN 2048
#define NH 16
#define HD 128
#define SEQ 2048
#define BATCH 2
#define INV_NORM 0.08838834764831845f  // 1/sqrt(128)

typedef __attribute__((ext_vector_type(16))) __bf16 v16bf;
typedef __attribute__((ext_vector_type(8)))  __bf16 v8bf;
typedef __attribute__((ext_vector_type(8)))  float  v8f;
typedef __attribute__((ext_vector_type(4)))  unsigned int v4u;
typedef int v4i_t __attribute__((vector_size(16)));  // matches builtin prototype

#if __has_builtin(__builtin_amdgcn_global_load_async_to_lds_b128) && \
    __has_builtin(__builtin_amdgcn_s_wait_asynccnt)
#define HAVE_ASYNC_LDS 1
#else
#define HAVE_ASYNC_LDS 0
#endif

// Copy 16 bytes global -> LDS. Async DMA (ASYNCcnt-tracked) when available.
__device__ __forceinline__ void copy16_g2l(const void* g, void* l) {
#if HAVE_ASYNC_LDS
  __builtin_amdgcn_global_load_async_to_lds_b128(
      (__attribute__((address_space(1))) v4i_t*)g,
      (__attribute__((address_space(3))) v4i_t*)l, 0, 0);
#else
  v4u d = *(const v4u*)g;
  *(v4u*)l = d;
#endif
}
__device__ __forceinline__ void copy_wait() {
#if HAVE_ASYNC_LDS
  __builtin_amdgcn_s_wait_asynccnt(0);
#endif
}

// Build a 16-element bf16 fragment from two 16-byte-aligned 8-element chunks.
__device__ __forceinline__ v16bf ld_frag(const __bf16* p0, const __bf16* p1) {
  v8bf lo = *(const v8bf*)p0;
  v8bf hi = *(const v8bf*)p1;
  v16bf r;
#pragma unroll
  for (int i = 0; i < 8; ++i) { r[i] = lo[i]; r[i + 8] = hi[i]; }
  return r;
}

// ---------------------------------------------------------------------------
// fp32 -> bf16 conversion (n must be a multiple of 8)
// ---------------------------------------------------------------------------
__global__ void cvt_f32_bf16(const float* __restrict__ src,
                             __bf16* __restrict__ dst, int n) {
  int i = (blockIdx.x * blockDim.x + threadIdx.x) * 8;
  if (i + 7 < n) {
    float4 f0 = *(const float4*)(src + i);
    float4 f1 = *(const float4*)(src + i + 4);
    v8bf o;
    o[0] = (__bf16)f0.x; o[1] = (__bf16)f0.y; o[2] = (__bf16)f0.z; o[3] = (__bf16)f0.w;
    o[4] = (__bf16)f1.x; o[5] = (__bf16)f1.y; o[6] = (__bf16)f1.z; o[7] = (__bf16)f1.w;
    *(v8bf*)(dst + i) = o;
  }
}

// ---------------------------------------------------------------------------
// WMMA GEMM: C[M,N] = A[M,K] * W[N,K]^T (+bias, +residual)
// N == K == HIDDEN. Block = 256 threads (8 waves), tile 128(M) x 128(N).
// Double-buffered async LDS staging of W tiles.
// mode 0/1: store bf16 as [B,NH,S,D] (Q,K)   mode 2: [B,NH,D,S] (V^T)
// mode 3: store fp32 row-major + bias + residual (output projection)
// ---------------------------------------------------------------------------
__global__ __launch_bounds__(256) void gemm_wmma(
    const __bf16* __restrict__ A, const __bf16* __restrict__ W,
    const float* __restrict__ bias, const float* __restrict__ resid,
    void* __restrict__ out, int mode) {
  __shared__ __bf16 Bt[2][128][40];  // ping-pong, 32 cols + 8 pad

  const int tid   = threadIdx.x;
  const int wave  = tid >> 5;
  const int lane  = tid & 31;
  const int lhalf = lane >> 4;
  const int l16   = lane & 15;
  const int m0    = blockIdx.y * 128 + wave * 16;
  const int n0    = blockIdx.x * 128;
  const int nrow  = tid >> 1, nhalf = tid & 1;

  v8f acc[8];
#pragma unroll
  for (int t = 0; t < 8; ++t)
#pragma unroll
    for (int i = 0; i < 8; ++i) acc[t][i] = 0.0f;

  const __bf16* arow = A + (size_t)(m0 + l16) * HIDDEN + lhalf * 8;
  const __bf16* wrow = W + (size_t)(n0 + nrow) * HIDDEN + nhalf * 16;

  auto stage_b = [&](int k0, int buf) {
    const __bf16* src = wrow + k0;
    __bf16* dst = &Bt[buf][nrow][nhalf * 16];
    copy16_g2l(src, dst);
    copy16_g2l(src + 8, dst + 8);
  };

  stage_b(0, 0);  // prologue
  for (int k0 = 0; k0 < HIDDEN; k0 += 32) {
    const int buf = (k0 >> 5) & 1;
    copy_wait();
    __syncthreads();
    if (k0 + 32 < HIDDEN) {
      stage_b(k0 + 32, buf ^ 1);  // overlap next tile's DMA with compute
      if (k0 + 64 < HIDDEN)
        __builtin_prefetch(wrow + k0 + 64, 0, 1);  // global_prefetch_b8
    }

    v16bf a = ld_frag(arow + k0, arow + k0 + 16);
    v16bf bfr[8];
#pragma unroll
    for (int t = 0; t < 8; ++t) {
      const __bf16* bp = &Bt[buf][t * 16 + l16][lhalf * 8];
      bfr[t] = ld_frag(bp, bp + 16);
    }
#pragma unroll
    for (int t = 0; t < 8; ++t)
      acc[t] = __builtin_amdgcn_wmma_f32_16x16x32_bf16(
          false, a, false, bfr[t], (short)0, acc[t], false, false);
  }

#pragma unroll
  for (int t = 0; t < 8; ++t) {
    const int ncol = n0 + t * 16 + l16;
    const float bv = bias[ncol];
#pragma unroll
    for (int r = 0; r < 8; ++r) {
      const int mrow = m0 + r + 8 * lhalf;
      const float v = acc[t][r] + bv;
      if (mode == 3) {
        const size_t idx = (size_t)mrow * HIDDEN + ncol;
        ((float*)out)[idx] = v + resid[idx];
      } else {
        const int b = mrow >> 11, s = mrow & 2047;
        const int h = ncol >> 7,  d = ncol & 127;
        size_t idx;
        if (mode == 2) idx = ((size_t)(b * NH + h) * HD + d) * SEQ + s;  // V^T
        else           idx = ((size_t)(b * NH + h) * SEQ + s) * HD + d;  // Q, K
        ((__bf16*)out)[idx] = (__bf16)v;
      }
    }
  }
}

// ---------------------------------------------------------------------------
// Flash attention: scores = alibi + inv_norm * Q K^T, softmax, ctx = P V.
// Block = 128 threads (4 waves); each wave owns 16 queries; grid (S/64, NH, B).
// Double-buffered async LDS staging of K / V^T tiles.
// ---------------------------------------------------------------------------
__global__ __launch_bounds__(128) void attn_wmma(
    const __bf16* __restrict__ Q,   // [B,NH,S,D]
    const __bf16* __restrict__ K,   // [B,NH,S,D]
    const __bf16* __restrict__ Vt,  // [B,NH,D,S]
    const float* __restrict__ alibi,// [B*NH, S]
    __bf16* __restrict__ ctx) {     // [B, S, HIDDEN]
  __shared__ __bf16 Kt[2][32][136];  // 32 keys x 128 d (+8 pad), ping-pong
  __shared__ __bf16 Vs[2][128][40];  // 128 d x 32 keys (+8 pad), ping-pong
  __shared__ __bf16 Pb[4][16][32];   // per-wave probs tile

  const int tid   = threadIdx.x;
  const int wave  = tid >> 5;
  const int lane  = tid & 31;
  const int lhalf = lane >> 4;
  const int l16   = lane & 15;
  const int b = blockIdx.z, h = blockIdx.y;
  const int q0 = blockIdx.x * 64 + wave * 16;

  const size_t head = (size_t)(b * NH + h) * SEQ * HD;
  const __bf16* Kh = K + head;
  const __bf16* Vh = Vt + head;
  const float*  al = alibi + (size_t)(b * NH + h) * SEQ;

  // Q fragments for the 4 contraction steps over D=128
  v16bf aq[4];
  {
    const __bf16* qrow = Q + head + (size_t)(q0 + l16) * HD + lhalf * 8;
#pragma unroll
    for (int ds = 0; ds < 4; ++ds)
      aq[ds] = ld_frag(qrow + ds * 32, qrow + ds * 32 + 16);
  }

  float mrow[8], lrow[8];
  v8f acc[8];
#pragma unroll
  for (int r = 0; r < 8; ++r) { mrow[r] = -1e30f; lrow[r] = 0.0f; }
#pragma unroll
  for (int t = 0; t < 8; ++t)
#pragma unroll
    for (int i = 0; i < 8; ++i) acc[t][i] = 0.0f;

  const int kkey = tid >> 2, kqtr = tid & 3;
  auto stage_kv = [&](int kt, int buf) {
    {  // 32 keys x 128 d of K
      const __bf16* src = Kh + (size_t)(kt + kkey) * HD + kqtr * 32;
      __bf16* dst = &Kt[buf][kkey][kqtr * 32];
      copy16_g2l(src, dst);
      copy16_g2l(src + 8, dst + 8);
    }
    {  // 128 d x 32 keys of V^T
      const __bf16* src = Vh + (size_t)tid * SEQ + kt;
      __bf16* dst = &Vs[buf][tid][0];
#pragma unroll
      for (int i = 0; i < 4; ++i) copy16_g2l(src + i * 8, dst + i * 8);
    }
  };

  stage_kv(0, 0);  // prologue
  for (int kt = 0; kt < SEQ; kt += 32) {
    const int buf = (kt >> 5) & 1;
    copy_wait();
    __syncthreads();
    if (kt + 32 < SEQ) stage_kv(kt + 32, buf ^ 1);

#pragma unroll
    for (int half = 0; half < 2; ++half) {
      v8f s;
#pragma unroll
      for (int i = 0; i < 8; ++i) s[i] = 0.0f;
      v16bf bk[4];
#pragma unroll
      for (int ds = 0; ds < 4; ++ds) {
        const __bf16* bp = &Kt[buf][half * 16 + l16][ds * 32 + lhalf * 8];
        bk[ds] = ld_frag(bp, bp + 16);
      }
#pragma unroll
      for (int ds = 0; ds < 4; ++ds)
        s = __builtin_amdgcn_wmma_f32_16x16x32_bf16(
            false, aq[ds], false, bk[ds], (short)0, s, false, false);

      const float av = al[kt + half * 16 + l16];
#pragma unroll
      for (int r = 0; r < 8; ++r) s[r] = s[r] * INV_NORM + av;

#pragma unroll
      for (int r = 0; r < 8; ++r) {
        float t = s[r];  // row max across the 16-lane group
        t = fmaxf(t, __shfl_xor(t, 1, 16));
        t = fmaxf(t, __shfl_xor(t, 2, 16));
        t = fmaxf(t, __shfl_xor(t, 4, 16));
        t = fmaxf(t, __shfl_xor(t, 8, 16));
        const float mnew = fmaxf(mrow[r], t);
        const float sc = __expf(mrow[r] - mnew);
        const float p  = __expf(s[r] - mnew);
        float ps = p;
        ps += __shfl_xor(ps, 1, 16);
        ps += __shfl_xor(ps, 2, 16);
        ps += __shfl_xor(ps, 4, 16);
        ps += __shfl_xor(ps, 8, 16);
        lrow[r] = lrow[r] * sc + ps;
        mrow[r] = mnew;
#pragma unroll
        for (int t2 = 0; t2 < 8; ++t2) acc[t2][r] *= sc;
        Pb[wave][r + 8 * lhalf][half * 16 + l16] = (__bf16)p;
      }
    }
    __syncthreads();

    // ctx += P(16x32) * V(32x16) per d-subtile
    const __bf16* prow = &Pb[wave][l16][lhalf * 8];
    v16bf ap = ld_frag(prow, prow + 16);
#pragma unroll
    for (int g = 0; g < 2; ++g) {
      v16bf bv[4];
#pragma unroll
      for (int t = 0; t < 4; ++t) {
        const __bf16* bp = &Vs[buf][(g * 4 + t) * 16 + l16][lhalf * 8];
        bv[t] = ld_frag(bp, bp + 16);
      }
#pragma unroll
      for (int t = 0; t < 4; ++t)
        acc[g * 4 + t] = __builtin_amdgcn_wmma_f32_16x16x32_bf16(
            false, ap, false, bv[t], (short)0, acc[g * 4 + t], false, false);
    }
  }

#pragma unroll
  for (int t = 0; t < 8; ++t) {
#pragma unroll
    for (int r = 0; r < 8; ++r) {
      const int qr = q0 + r + 8 * lhalf;
      const int d  = t * 16 + l16;
      const float v = acc[t][r] / lrow[r];
      ctx[((size_t)b * SEQ + qr) * HIDDEN + h * HD + d] = (__bf16)v;
    }
  }
}

// ---------------------------------------------------------------------------
extern "C" void kernel_launch(void* const* d_in, const int* in_sizes, int n_in,
                              void* d_out, int out_size, void* d_ws, size_t ws_size,
                              hipStream_t stream) {
  const float* hidden   = (const float*)d_in[0];
  const float* residual = (const float*)d_in[1];
  const float* alibi    = (const float*)d_in[2];
  const float* Wq = (const float*)d_in[3];
  const float* bq = (const float*)d_in[4];
  const float* Wk = (const float*)d_in[5];
  const float* bk = (const float*)d_in[6];
  const float* Wv = (const float*)d_in[7];
  const float* bv = (const float*)d_in[8];
  const float* Wd = (const float*)d_in[9];
  const float* bd = (const float*)d_in[10];

  const size_t nAct = (size_t)BATCH * SEQ * HIDDEN;  // 8388608
  const size_t nW   = (size_t)HIDDEN * HIDDEN;       // 4194304

  char* ws = (char*)d_ws;
  __bf16* Xb  = (__bf16*)ws; ws += nAct * 2;
  __bf16* Wqb = (__bf16*)ws; ws += nW * 2;
  __bf16* Wkb = (__bf16*)ws; ws += nW * 2;
  __bf16* Wvb = (__bf16*)ws; ws += nW * 2;
  __bf16* Wdb = (__bf16*)ws; ws += nW * 2;
  __bf16* Qb  = (__bf16*)ws; ws += nAct * 2;
  __bf16* Kb  = (__bf16*)ws; ws += nAct * 2;
  __bf16* Vtb = (__bf16*)ws; ws += nAct * 2;
  __bf16* Cb  = (__bf16*)ws; ws += nAct * 2;

  // 1) fp32 -> bf16 conversions
  cvt_f32_bf16<<<(int)(nAct / (256 * 8)), 256, 0, stream>>>(hidden, Xb, (int)nAct);
  cvt_f32_bf16<<<(int)(nW / (256 * 8)), 256, 0, stream>>>(Wq, Wqb, (int)nW);
  cvt_f32_bf16<<<(int)(nW / (256 * 8)), 256, 0, stream>>>(Wk, Wkb, (int)nW);
  cvt_f32_bf16<<<(int)(nW / (256 * 8)), 256, 0, stream>>>(Wv, Wvb, (int)nW);
  cvt_f32_bf16<<<(int)(nW / (256 * 8)), 256, 0, stream>>>(Wd, Wdb, (int)nW);

  // 2) QKV projections (M = B*S = 4096 rows)
  dim3 ggrid(HIDDEN / 128, (BATCH * SEQ) / 128);  // (16, 32)
  gemm_wmma<<<ggrid, 256, 0, stream>>>(Xb, Wqb, bq, nullptr, Qb, 0);
  gemm_wmma<<<ggrid, 256, 0, stream>>>(Xb, Wkb, bk, nullptr, Kb, 1);
  gemm_wmma<<<ggrid, 256, 0, stream>>>(Xb, Wvb, bv, nullptr, Vtb, 2);

  // 3) attention
  attn_wmma<<<dim3(SEQ / 64, NH, BATCH), 128, 0, stream>>>(Qb, Kb, Vtb, alibi, Cb);

  // 4) output projection + bias + residual (fp32 output)
  gemm_wmma<<<ggrid, 256, 0, stream>>>(Cb, Wdb, bd, residual, d_out, 3);
}
